// MultiHeadSSM_21217138442323
// MI455X (gfx1250) — matrix operand order repaired and checked by reading, not compiled, not run on hardware
//
#include <hip/hip_runtime.h>

// ---------------- problem constants ----------------
#define BATCH   4
#define TLEN    4096
#define DMODEL  2048
#define NHEADS  32
#define HEADDIM 64
#define BTROWS  (BATCH * TLEN)          // 16384
#define NCHUNK  64
#define CHLEN   (TLEN / NCHUNK)         // 64
#define KSTAGE  64

typedef __bf16 v16bf __attribute__((ext_vector_type(16)));
typedef float  v8f   __attribute__((ext_vector_type(8)));
typedef unsigned int u32x4 __attribute__((ext_vector_type(4)));
typedef unsigned int u32x8 __attribute__((ext_vector_type(8)));

// LDS staging geometry (padded by TDM pad feature)
#define A_ROWB   144                     // 128B row + 16B pad  (pad_interval=32dw, amount=4dw)
#define B_ROWB   272                     // 256B row + 16B pad  (pad_interval=64dw, amount=4dw)
#define A_BUFB   (128 * A_ROWB)          // 18432
#define B_BUFB   (64 * B_ROWB)           // 17408
#define SA_OFF(b) ((b) * A_BUFB)
#define SB_OFF(b) (2 * A_BUFB + (b) * B_BUFB)
#define LDS_TOT  (2 * A_BUFB + 2 * B_BUFB)

__device__ __forceinline__ unsigned short f2bf(float f) {
  unsigned int u = __float_as_uint(f);
  u += 0x7FFFu + ((u >> 16) & 1u);          // round-to-nearest-even
  return (unsigned short)(u >> 16);
}

__device__ __forceinline__ v16bf pack_frag(uint4 lo, uint4 hi) {
  union { unsigned int u[8]; v16bf v; } t;
  t.u[0] = lo.x; t.u[1] = lo.y; t.u[2] = lo.z; t.u[3] = lo.w;
  t.u[4] = hi.x; t.u[5] = hi.y; t.u[6] = hi.z; t.u[7] = hi.w;
  return t.v;
}

// TDM issue: 2D descriptor (groups 0+1 only, VADDR2/3 = NULL)
__device__ __forceinline__ void tdm_load2(u32x4 g0, u32x8 g1) {
#if defined(__AMDGCN__)
  asm volatile("tensor_load_to_lds %0, %1" :: "s"(g0), "s"(g1) : "memory");
#else
  (void)g0; (void)g1;
#endif
}

// ---------------- fp32 -> bf16 bulk convert (float4 vectorized) ----------------
__global__ __launch_bounds__(256)
void ssm_cvt_bf16_k(const float* __restrict__ in, unsigned short* __restrict__ out, long n4) {
  long i = (long)blockIdx.x * blockDim.x + threadIdx.x;
  long stride = (long)gridDim.x * blockDim.x;
  for (; i < n4; i += stride) {
    float4 v = ((const float4*)in)[i];
    ushort4 o;
    o.x = f2bf(v.x); o.y = f2bf(v.y); o.z = f2bf(v.z); o.w = f2bf(v.w);
    ((ushort4*)out)[i] = o;
  }
}

// ---------------- W (N x K, f32 row-major) -> Wt (K x N, bf16) via LDS tile ----------------
__global__ __launch_bounds__(256)
void ssm_transW_k(const float* __restrict__ W, unsigned short* __restrict__ Wt, int N, int K) {
  __shared__ unsigned short tile[32][33];
  const int kb = blockIdx.x * 32;
  const int nb = blockIdx.y * 32;
  const int tx = threadIdx.x & 31;
  const int ty = threadIdx.x >> 5;            // 8 rows per pass
  #pragma unroll
  for (int r = ty; r < 32; r += 8)
    tile[r][tx] = f2bf(W[(size_t)(nb + r) * K + kb + tx]);   // tile[n_local][k_local]
  __syncthreads();
  #pragma unroll
  for (int r = ty; r < 32; r += 8)
    Wt[(size_t)(kb + r) * N + nb + tx] = tile[tx][r];
}

// ---------------- bf16 WMMA GEMM, TDM double-buffered through LDS ----------------
// C[M,N] = A[M,K] * Wt[K,N]; block 256 = 8 waves; 128x128 tile; wave = 64(M) x 32(N).
// Wave 0 DMAs the next K-stage (A 128x64, B 64x128 bf16 tiles) with tensor_load_to_lds
// while all waves compute WMMAs from the current LDS buffers.
__global__ __launch_bounds__(256)
void ssm_gemm_bf16_k(const unsigned short* __restrict__ A,
                     const unsigned short* __restrict__ Wt,
                     float* __restrict__ C, int M, int N, int K) {
  __shared__ __align__(16) unsigned char lds[LDS_TOT];

  const int lane = threadIdx.x & 31;
  const int wave = threadIdx.x >> 5;
  const int mB = blockIdx.y * 128;
  const int nB = blockIdx.x * 128;
  const int wm = (wave & 1) * 64;             // wave M offset in tile
  const int wn = (wave >> 1) * 32;            // wave N offset in tile
  const int rowsel = lane & 15;
  const int kh = (lane >> 4) * 8;             // A-frag K-half per ISA layout

  v8f acc[4][2];
  #pragma unroll
  for (int i = 0; i < 4; i++)
    #pragma unroll
    for (int j = 0; j < 2; j++) { v8f z = {}; acc[i][j] = z; }

  // ---- D# group1 descriptors (wave-uniform constants) ----
  // A tile: dim0 = K (contig), dim1 = M rows; tile 64 x 128; stride0 = K.
  // word0: data_size=1(2B)<<16 | pad_en<<20 | pad_interval<<22 | pad_amount<<25
  u32x8 g1a;
  g1a[0] = 0x00010000u | (1u << 20) | (4u << 22) | (3u << 25);     // pad every 32dw by 4dw
  g1a[1] = ((unsigned)K & 0xFFFFu) << 16;
  g1a[2] = (((unsigned)K >> 16) & 0xFFFFu) | (((unsigned)M & 0xFFFFu) << 16);
  g1a[3] = (((unsigned)M >> 16) & 0xFFFFu) | (64u << 16);          // tile_dim0 = 64
  g1a[4] = 128u;                                                    // tile_dim1 = 128
  g1a[5] = (unsigned)K;                                             // stride0
  g1a[6] = 0u; g1a[7] = 0u;
  // B tile: dim0 = N (contig), dim1 = K rows; tile 128 x 64; stride0 = N.
  u32x8 g1b;
  g1b[0] = 0x00010000u | (1u << 20) | (5u << 22) | (3u << 25);     // pad every 64dw by 4dw
  g1b[1] = ((unsigned)N & 0xFFFFu) << 16;
  g1b[2] = (((unsigned)N >> 16) & 0xFFFFu) | (((unsigned)K & 0xFFFFu) << 16);
  g1b[3] = (((unsigned)K >> 16) & 0xFFFFu) | (128u << 16);         // tile_dim0 = 128
  g1b[4] = 64u;                                                     // tile_dim1 = 64
  g1b[5] = (unsigned)N;                                             // stride0
  g1b[6] = 0u; g1b[7] = 0u;

  const int nstage = K / KSTAGE;

  auto issue = [&](int stage, int buf) {
    if (wave == 0) {
      unsigned long long ga = (unsigned long long)(size_t)A +
                              ((size_t)mB * K + (size_t)stage * KSTAGE) * 2u;
      u32x4 g0;
      g0[0] = 1u;                                                   // count=1, user mode
      g0[1] = (unsigned)(size_t)&lds[SA_OFF(buf)];
      g0[2] = (unsigned)ga;
      g0[3] = (unsigned)(ga >> 32) | 0x80000000u;                   // type=2 (bits 127:126)
      tdm_load2(g0, g1a);
      unsigned long long gb = (unsigned long long)(size_t)Wt +
                              ((size_t)stage * KSTAGE * N + (size_t)nB) * 2u;
      u32x4 h0;
      h0[0] = 1u;
      h0[1] = (unsigned)(size_t)&lds[SB_OFF(buf)];
      h0[2] = (unsigned)gb;
      h0[3] = (unsigned)(gb >> 32) | 0x80000000u;
      tdm_load2(h0, g1b);
    }
  };

  issue(0, 0);
  for (int st = 0; st < nstage; ++st) {
    const int buf = st & 1;
    if (st + 1 < nstage) {
      issue(st + 1, buf ^ 1);
      if (wave == 0) __builtin_amdgcn_s_wait_tensorcnt(2);   // stage st complete (in-order)
    } else {
      if (wave == 0) __builtin_amdgcn_s_wait_tensorcnt(0);
    }
    __syncthreads();                                         // stage st visible to all waves

    #pragma unroll
    for (int kk2 = 0; kk2 < KSTAGE; kk2 += 32) {
      v16bf af[4], bfr[2];
      #pragma unroll
      for (int i = 0; i < 4; i++) {
        const uint4* pa = (const uint4*)&lds[SA_OFF(buf) +
                            (wm + i * 16 + rowsel) * A_ROWB + (kk2 + kh) * 2];
        af[i] = pack_frag(pa[0], pa[2]);                     // K halves +0 / +16 elems
      }
      #pragma unroll
      for (int j = 0; j < 2; j++) {
        const unsigned short* pb = (const unsigned short*)&lds[SB_OFF(buf) +
                            (kk2 + kh) * B_ROWB + (wn + j * 16 + rowsel) * 2];
        union { unsigned short u[16]; v16bf v; } tb;
        #pragma unroll
        for (int s = 0; s < 8; s++) {
          tb.u[s] = pb[s * (B_ROWB / 2)];
          tb.u[8 + s] = pb[(16 + s) * (B_ROWB / 2)];
        }
        bfr[j] = tb.v;                                       // column n = lane%16, K halves +0 / +16 elems
      }
      #pragma unroll
      for (int i = 0; i < 4; i++)
        #pragma unroll
        for (int j = 0; j < 2; j++)
          acc[i][j] = __builtin_amdgcn_wmma_f32_16x16x32_bf16(
              false, af[i], false, bfr[j], (short)0, acc[i][j], false, false);
    }
    __syncthreads();                                         // done reading buf before reuse
  }

  const int cn = lane & 15;
  const int mo = (lane >> 4) * 8;
  #pragma unroll
  for (int i = 0; i < 4; i++)
    #pragma unroll
    for (int j = 0; j < 2; j++)
      #pragma unroll
      for (int r = 0; r < 8; r++)
        C[(size_t)(mB + wm + i * 16 + mo + r) * N + (nB + wn + j * 16 + cn)] = acc[i][j][r];
}

// ---------------- dt = softplus(xp @ W_dt^T + b_dt), one block per (b,t) row ----------------
// Row staged into LDS with async global->LDS copies (ASYNCcnt path).
__global__ __launch_bounds__(256)
void ssm_dt_k(const float* __restrict__ xp, const float* __restrict__ W_dt,
              const float* __restrict__ b_dt, float* __restrict__ dt) {
  __shared__ __align__(16) float row[DMODEL];
  const int bt = blockIdx.x;
  const float* xr = xp + (size_t)bt * DMODEL;
#if defined(__AMDGCN__)
  {
    const float* g0 = xr + threadIdx.x * 4;
    const float* g1 = g0 + 1024;
    unsigned l0 = (unsigned)(size_t)&row[threadIdx.x * 4];
    unsigned l1 = (unsigned)(size_t)&row[threadIdx.x * 4 + 1024];
    asm volatile("global_load_async_to_lds_b128 %0, %1, off" :: "v"(l0), "v"(g0) : "memory");
    asm volatile("global_load_async_to_lds_b128 %0, %1, off" :: "v"(l1), "v"(g1) : "memory");
    asm volatile("s_wait_asynccnt 0x0" ::: "memory");
  }
#else
  for (int i = threadIdx.x; i < DMODEL; i += 256) row[i] = xr[i];
#endif
  __syncthreads();
  const int wave = threadIdx.x >> 5;
  const int lane = threadIdx.x & 31;
  for (int h = wave; h < NHEADS; h += 8) {
    const float* w = W_dt + (size_t)h * DMODEL;
    float s = 0.f;
    for (int d = lane; d < DMODEL; d += 32) s += row[d] * w[d];
    #pragma unroll
    for (int off = 16; off > 0; off >>= 1) s += __shfl_xor(s, off, 32);
    if (lane == 0) {
      float v = s + b_dt[h];
      dt[(size_t)bt * NHEADS + h] = (v > 20.f) ? v : log1pf(__expf(v));
    }
  }
}

// ---------------- scan pass 1: per-chunk (prod a, local u) ----------------
__global__ __launch_bounds__(256)
void ssm_scan1_k(const float* __restrict__ xp, const float* __restrict__ dt,
                 const float* __restrict__ log_A,
                 float* __restrict__ chA, float* __restrict__ chU) {
  const int d = threadIdx.x;
  const int h = blockIdx.y * 4 + threadIdx.y;
  const int c = blockIdx.x;
  const int b = blockIdx.z;
  const float Ac = -__expf(log_A[h * HEADDIM + d]);
  const size_t xbase = ((size_t)b * TLEN) * DMODEL + h * HEADDIM + d;
  float ap = 1.f, u = 0.f;
  for (int i = 0; i < CHLEN; i++) {
    const int t = c * CHLEN + i;
    const float dv = dt[((size_t)b * TLEN + t) * NHEADS + h];
    const float xv = xp[xbase + (size_t)t * DMODEL];
    const float a = __expf(dv * Ac);
    u = a * u + xv;
    ap *= a;
  }
  const size_t ch = (((size_t)b * NHEADS + h) * HEADDIM + d) * NCHUNK + c;
  chA[ch] = ap; chU[ch] = u;
}

// ---------------- scan pass 2: serial prefix over chunks; init states + final s ----------------
__global__ __launch_bounds__(256)
void ssm_scan2_k(const float* __restrict__ chA, const float* __restrict__ chU,
                 float* __restrict__ chInit, float* __restrict__ s_out) {
  const int idx = blockIdx.x * 256 + threadIdx.x;   // (b*H + h)*64 + d == b*2048 + h*64 + d
  const float* pa = chA + (size_t)idx * NCHUNK;
  const float* pu = chU + (size_t)idx * NCHUNK;
  float* pi = chInit + (size_t)idx * NCHUNK;
  float s = 0.f;
  for (int c = 0; c < NCHUNK; c++) {
    pi[c] = s;
    s = pa[c] * s + pu[c];
  }
  s_out[idx] = s;
}

// ---------------- scan pass 3: replay chunks; ys in place + fused bf16 ----------------
__global__ __launch_bounds__(256)
void ssm_scan3_k(float* __restrict__ xp_ys, const float* __restrict__ dt,
                 const float* __restrict__ log_A, const float* __restrict__ chInit,
                 unsigned short* __restrict__ ys_bf) {
  const int d = threadIdx.x;
  const int h = blockIdx.y * 4 + threadIdx.y;
  const int c = blockIdx.x;
  const int b = blockIdx.z;
  const float Ac = -__expf(log_A[h * HEADDIM + d]);
  const size_t xbase = ((size_t)b * TLEN) * DMODEL + h * HEADDIM + d;
  float st = chInit[(((size_t)b * NHEADS + h) * HEADDIM + d) * NCHUNK + c];
  for (int i = 0; i < CHLEN; i++) {
    const int t = c * CHLEN + i;
    const size_t pos = xbase + (size_t)t * DMODEL;
    const float dv = dt[((size_t)b * TLEN + t) * NHEADS + h];
    const float a = __expf(dv * Ac);
    st = a * st + xp_ys[pos];
    xp_ys[pos] = st;
    ys_bf[pos] = f2bf(st);
  }
}

// ---------------- launcher ----------------
extern "C" void kernel_launch(void* const* d_in, const int* in_sizes, int n_in,
                              void* d_out, int out_size, void* d_ws, size_t ws_size,
                              hipStream_t stream) {
  (void)in_sizes; (void)n_in; (void)out_size; (void)ws_size;
  const float* x     = (const float*)d_in[0];
  const float* W_in  = (const float*)d_in[1];
  const float* W_out = (const float*)d_in[2];
  const float* log_A = (const float*)d_in[3];
  const float* W_dt  = (const float*)d_in[4];
  const float* b_dt  = (const float*)d_in[5];

  float* y     = (float*)d_out;                                   // BT*D
  float* s_out = (float*)d_out + (size_t)BTROWS * DMODEL;         // B*H*hd

  char* ws = (char*)d_ws;
  float*          xp     = (float*)(ws + 0);                      // 134,217,728 B
  unsigned short* bfA    = (unsigned short*)(ws + 134217728);     //  67,108,864 B
  unsigned short* bfW    = (unsigned short*)(ws + 201326592);     //   8,388,608 B
  float*          dtbuf  = (float*)(ws + 209715200);              //   2,097,152 B
  float*          chA    = (float*)(ws + 211812352);              //   2,097,152 B
  float*          chU    = (float*)(ws + 213909504);              //   2,097,152 B
  float*          chInit = (float*)(ws + 216006656);              //   2,097,152 B

  const long nXp4 = (long)BTROWS * DMODEL / 4;

  ssm_cvt_bf16_k<<<4096, 256, 0, stream>>>(x, bfA, nXp4);
  ssm_transW_k<<<dim3(DMODEL / 32, DMODEL / 32), 256, 0, stream>>>(W_in, bfW, DMODEL, DMODEL);
  ssm_gemm_bf16_k<<<dim3(DMODEL / 128, BTROWS / 128), 256, 0, stream>>>(
      bfA, bfW, xp, BTROWS, DMODEL, DMODEL);
  ssm_dt_k<<<BTROWS, 256, 0, stream>>>(xp, W_dt, b_dt, dtbuf);
  ssm_scan1_k<<<dim3(NCHUNK, NHEADS / 4, BATCH), dim3(64, 4), 0, stream>>>(xp, dtbuf, log_A, chA, chU);
  ssm_scan2_k<<<(BATCH * NHEADS * HEADDIM) / 256, 256, 0, stream>>>(chA, chU, chInit, s_out);
  ssm_scan3_k<<<dim3(NCHUNK, NHEADS / 4, BATCH), dim3(64, 4), 0, stream>>>(xp, dtbuf, log_A, chInit, bfA);
  ssm_transW_k<<<dim3(DMODEL / 32, DMODEL / 32), 256, 0, stream>>>(W_out, bfW, DMODEL, DMODEL);
  ssm_gemm_bf16_k<<<dim3(DMODEL / 128, BTROWS / 128), 256, 0, stream>>>(
      bfA, bfW, y, BTROWS, DMODEL, DMODEL);
}
